// SkipGPTAttention_8306466750647
// MI455X (gfx1250) — matrix operand order repaired and checked by reading, not compiled, not run on hardware
//
#include <hip/hip_runtime.h>
#include <hip/hip_bf16.h>

// ---------------------------------------------------------------------------
// SkipGPT attention layer for gfx1250 (MI455X), bf16 WMMA everywhere.
// f32 in/out, bf16 WMMA with f32 accumulation.  LDS tiles held in bf16 so the
// inner loops are pure ds_load_b128 -> v_wmma; f32->bf16 conversion happens
// once per staged element, overlapped with compute via register double-buffer.
// ---------------------------------------------------------------------------

typedef __attribute__((ext_vector_type(16))) __bf16 v16bf;
typedef __attribute__((ext_vector_type(8)))  __bf16 bf16x8;
typedef __attribute__((ext_vector_type(4)))  __bf16 bf16x4;
typedef __attribute__((ext_vector_type(8)))  float  v8f;
typedef __attribute__((ext_vector_type(4)))  float  f32x4;

#define S_LEN   2048
#define HQ_N    32
#define HKV_N   8
#define D_N     128
#define HID_N   4096
#define SCALE_F 0.08838834764831845f   // 128^-0.5

static __device__ __forceinline__ v8f wmma_bf16(v16bf a, v16bf b, v8f c) {
  return __builtin_amdgcn_wmma_f32_16x16x32_bf16(
      false, a, false, b, (short)0, c, false, false);
}

static __device__ __forceinline__ v16bf cat8(bf16x8 lo, bf16x8 hi) {
  return __builtin_shufflevector(lo, hi, 0, 1, 2, 3, 4, 5, 6, 7,
                                 8, 9, 10, 11, 12, 13, 14, 15);
}

static __device__ __forceinline__ bf16x4 cvt4(f32x4 v) {
  bf16x4 o;
  o[0] = (__bf16)v[0]; o[1] = (__bf16)v[1];
  o[2] = (__bf16)v[2]; o[3] = (__bf16)v[3];
  return o;
}

// ---------------------------------------------------------------------------
// C[M,N] = A[M,K] @ W[N,K]^T, f32 in/out, bf16 WMMA compute.
// Block: 256 threads = 8 waves.  Block tile 128x128, K-tile 64 (2 WMMA steps).
// Wave tile 64x32 (4x2 WMMA tiles).  LDS double-buffered in bf16.
// Optional rowMask multiplies each output row by (float)rowMask[row].
// ---------------------------------------------------------------------------
__global__ __launch_bounds__(256)
void gemm_xwT_bf16wmma(const float* __restrict__ A, const float* __restrict__ W,
                       float* __restrict__ C, int M, int N, int K,
                       const int* __restrict__ rowMask)
{
  __shared__ __align__(16) __bf16 As[2][128][72];  // 64 cols + 8 pad (16B-mult)
  __shared__ __align__(16) __bf16 Bs[2][128][72];

  const int tid  = threadIdx.x;
  const int wave = tid >> 5;
  const int lane = tid & 31;
  const int half = lane >> 4;
  const int l16  = lane & 15;
  const int wm   = wave >> 2;     // 0..1 -> row offset wm*64
  const int wn   = wave & 3;      // 0..3 -> col offset wn*32
  const int m0   = blockIdx.y * 128;
  const int n0   = blockIdx.x * 128;
  const int nK   = K >> 6;        // 64-wide K tiles

  v8f acc[4][2];
  #pragma unroll
  for (int mt = 0; mt < 4; ++mt)
    #pragma unroll
    for (int nt = 0; nt < 2; ++nt)
      #pragma unroll
      for (int e = 0; e < 8; ++e) acc[mt][nt][e] = 0.0f;

  // ---- prologue: stage K-tile 0 (global f32 -> cvt -> LDS bf16) ----
  #pragma unroll
  for (int i = 0; i < 8; ++i) {
    int idx = tid + i * 256;            // 0..2047 float4 slots
    int r   = idx >> 4;                 // 0..127
    int c4  = idx & 15;                 // 0..15
    f32x4 a = *(const f32x4*)(A + (size_t)(m0 + r) * K + c4 * 4);
    f32x4 b = *(const f32x4*)(W + (size_t)(n0 + r) * K + c4 * 4);
    *(bf16x4*)(&As[0][r][c4 * 4]) = cvt4(a);
    *(bf16x4*)(&Bs[0][r][c4 * 4]) = cvt4(b);
  }
  __syncthreads();

  for (int kt = 0; kt < nK; ++kt) {
    const int cur = kt & 1, nxt = cur ^ 1;
    const bool have_next = (kt + 1) < nK;

    // ---- issue global loads for tile kt+1 early (overlap with WMMA) ----
    f32x4 regA[8], regB[8];
    if (have_next) {
      const int k0 = (kt + 1) * 64;
      #pragma unroll
      for (int i = 0; i < 8; ++i) {
        int idx = tid + i * 256;
        int r   = idx >> 4;
        int c4  = idx & 15;
        regA[i] = *(const f32x4*)(A + (size_t)(m0 + r) * K + k0 + c4 * 4);
        regB[i] = *(const f32x4*)(W + (size_t)(n0 + r) * K + k0 + c4 * 4);
      }
      if (kt + 2 < nK) {  // L2 hint for the tile after next (global_prefetch)
        __builtin_prefetch(A + (size_t)(m0 + (tid >> 1)) * K + (kt + 2) * 64, 0, 2);
        __builtin_prefetch(W + (size_t)(n0 + (tid >> 1)) * K + (kt + 2) * 64, 0, 2);
      }
    }

    // ---- compute: 2 WMMA k-steps on buffer `cur`, pure ds_load_b128 feed ----
    #pragma unroll
    for (int ks = 0; ks < 2; ++ks) {
      v16bf bfrag[2];
      #pragma unroll
      for (int nt = 0; nt < 2; ++nt) {
        const __bf16* row = &Bs[cur][wn * 32 + nt * 16 + l16][ks * 32];
        bfrag[nt] = cat8(*(const bf16x8*)(row + 8 * half),
                         *(const bf16x8*)(row + 16 + 8 * half));
      }
      v16bf afrag[4];
      #pragma unroll
      for (int mt = 0; mt < 4; ++mt) {
        const __bf16* row = &As[cur][wm * 64 + mt * 16 + l16][ks * 32];
        afrag[mt] = cat8(*(const bf16x8*)(row + half * 8),
                         *(const bf16x8*)(row + 16 + half * 8));
      }
      #pragma unroll
      for (int mt = 0; mt < 4; ++mt)
        #pragma unroll
        for (int nt = 0; nt < 2; ++nt)
          acc[mt][nt] = wmma_bf16(afrag[mt], bfrag[nt], acc[mt][nt]);
    }

    // ---- convert + store tile kt+1 into the other buffer ----
    if (have_next) {
      #pragma unroll
      for (int i = 0; i < 8; ++i) {
        int idx = tid + i * 256;
        int r   = idx >> 4;
        int c4  = idx & 15;
        *(bf16x4*)(&As[nxt][r][c4 * 4]) = cvt4(regA[i]);
        *(bf16x4*)(&Bs[nxt][r][c4 * 4]) = cvt4(regB[i]);
      }
    }
    __syncthreads();
  }

  // ---- store: C tile 16x16, VGPR r -> row r + 8*half, col l16 ----
  #pragma unroll
  for (int mt = 0; mt < 4; ++mt) {
    #pragma unroll
    for (int r = 0; r < 8; ++r) {
      int row = m0 + wm * 64 + mt * 16 + r + 8 * half;
      float scale = rowMask ? (float)rowMask[row] : 1.0f;
      #pragma unroll
      for (int nt = 0; nt < 2; ++nt) {
        int col = n0 + wn * 32 + nt * 16 + l16;
        C[(size_t)row * N + col] = acc[mt][nt][r] * scale;
      }
    }
  }
}

// ---------------------------------------------------------------------------
// Per-(token, head) RMSNorm + RoPE, in place.  buf layout [s][H][128].
// ---------------------------------------------------------------------------
__global__ __launch_bounds__(128)
void rmsnorm_rope(float* __restrict__ buf, const float* __restrict__ w,
                  const float* __restrict__ cosb, const float* __restrict__ sinb,
                  int H)
{
  const int blk = blockIdx.x;
  const int s = blk / H;
  const int h = blk % H;
  const int d = threadIdx.x;
  float* x = buf + ((size_t)s * H + h) * D_N;

  __shared__ float red[128];
  __shared__ float ybuf[128];

  float xv = x[d];
  red[d] = xv * xv;
  __syncthreads();
  #pragma unroll
  for (int off = 64; off > 0; off >>= 1) {
    if (d < off) red[d] += red[d + off];
    __syncthreads();
  }
  float inv = rsqrtf(red[0] * (1.0f / (float)D_N) + 1e-6f);
  float y = xv * inv * w[d];
  ybuf[d] = y;
  __syncthreads();
  float rot = (d < 64) ? -ybuf[d + 64] : ybuf[d - 64];
  x[d] = y * cosb[(size_t)s * D_N + d] + rot * sinb[(size_t)s * D_N + d];
}

// ---------------------------------------------------------------------------
// Flash attention, GQA (4 Q heads per KV head), causal, bf16 WMMA.
// Grid: (HQ, S/64).  Block: 128 threads = 4 waves; each wave owns 16 query
// rows and the full D=128 output.  K/V staged to LDS as bf16 per 64-key block.
// ---------------------------------------------------------------------------
__global__ __launch_bounds__(128)
void flash_attn(const float* __restrict__ qb, const float* __restrict__ kb,
                const float* __restrict__ vb, float* __restrict__ ob)
{
  const int h    = blockIdx.x;          // query head
  const int hk   = h / (HQ_N / HKV_N);  // kv head
  const int qblk = blockIdx.y;          // 64-row query block
  const int tid  = threadIdx.x;
  const int wave = tid >> 5;
  const int lane = tid & 31;
  const int half = lane >> 4;
  const int l16  = lane & 15;
  const int q0   = qblk * 64 + wave * 16;

  __shared__ __align__(16) __bf16 Ks[64][D_N + 8];    // [key][d]
  __shared__ __align__(16) __bf16 Vt[D_N][64 + 8];    // [d][key]
  __shared__ __align__(16) __bf16 Ps[4][16][64 + 8];  // per-wave P slab

  // Q fragments: 16 rows x 128, scale folded in.  4 chunks of K=32.
  v16bf qf[4];
  {
    const float* qrow = qb + ((size_t)(q0 + l16) * HQ_N + h) * D_N;
    #pragma unroll
    for (int c = 0; c < 4; ++c) {
      #pragma unroll
      for (int j = 0; j < 8; ++j)
        qf[c][j] = (__bf16)(qrow[c * 32 + half * 8 + j] * SCALE_F);
      #pragma unroll
      for (int j = 0; j < 8; ++j)
        qf[c][8 + j] = (__bf16)(qrow[c * 32 + 16 + half * 8 + j] * SCALE_F);
    }
  }

  v8f oacc[8];
  #pragma unroll
  for (int dt = 0; dt < 8; ++dt)
    #pragma unroll
    for (int e = 0; e < 8; ++e) oacc[dt][e] = 0.0f;
  float rmax[8], rsum[8];
  #pragma unroll
  for (int r = 0; r < 8; ++r) { rmax[r] = -3.0e38f; rsum[r] = 0.0f; }

  for (int kbi = 0; kbi <= qblk; ++kbi) {
    const int kbase = kbi * 64;

    // ---- stage K (vectorized) and V (transposed) as bf16 ----
    #pragma unroll
    for (int i = 0; i < 16; ++i) {
      int idx = i * 128 + tid;          // 0..2047 float4 slots of K tile
      int key = idx >> 5;               // 0..63
      int c4  = idx & 31;               // d = c4*4
      f32x4 kv = *(const f32x4*)(kb + ((size_t)(kbase + key) * HKV_N + hk) * D_N + c4 * 4);
      *(bf16x4*)(&Ks[key][c4 * 4]) = cvt4(kv);
      f32x4 vv = *(const f32x4*)(vb + ((size_t)(kbase + key) * HKV_N + hk) * D_N + c4 * 4);
      #pragma unroll
      for (int j = 0; j < 4; ++j) Vt[c4 * 4 + j][key] = (__bf16)vv[j];
    }
    __syncthreads();

    // ---- S = Q @ K^T : 4 key tiles x 4 K-dim chunks ----
    v8f sacc[4];
    #pragma unroll
    for (int nt = 0; nt < 4; ++nt)
      #pragma unroll
      for (int e = 0; e < 8; ++e) sacc[nt][e] = 0.0f;
    #pragma unroll
    for (int nt = 0; nt < 4; ++nt) {
      #pragma unroll
      for (int c = 0; c < 4; ++c) {
        const __bf16* row = &Ks[nt * 16 + l16][c * 32 + 8 * half];
        v16bf bf_ = cat8(*(const bf16x8*)(row), *(const bf16x8*)(row + 16));
        sacc[nt] = wmma_bf16(qf[c], bf_, sacc[nt]);
      }
    }

    // ---- causal mask on the diagonal block ----
    if (kbi == qblk) {
      #pragma unroll
      for (int nt = 0; nt < 4; ++nt)
        #pragma unroll
        for (int r = 0; r < 8; ++r)
          if (nt * 16 + l16 > wave * 16 + r + 8 * half)
            sacc[nt][r] = -1.0e30f;
    }

    // ---- online softmax (rows spread across 16 lanes sharing `half`) ----
    #pragma unroll
    for (int r = 0; r < 8; ++r) {
      float mloc = sacc[0][r];
      #pragma unroll
      for (int nt = 1; nt < 4; ++nt) mloc = fmaxf(mloc, sacc[nt][r]);
      #pragma unroll
      for (int off = 1; off < 16; off <<= 1)
        mloc = fmaxf(mloc, __shfl_xor(mloc, off, 16));
      float mnew = fmaxf(rmax[r], mloc);
      float corr = __expf(rmax[r] - mnew);
      rmax[r] = mnew;
      rsum[r] *= corr;
      #pragma unroll
      for (int dt = 0; dt < 8; ++dt) oacc[dt][r] *= corr;
      float psum = 0.0f;
      #pragma unroll
      for (int nt = 0; nt < 4; ++nt) {
        float p = __expf(sacc[nt][r] - mnew);
        sacc[nt][r] = p;
        psum += p;
      }
      #pragma unroll
      for (int off = 1; off < 16; off <<= 1)
        psum += __shfl_xor(psum, off, 16);
      rsum[r] += psum;
    }

    // ---- relayout P (C layout -> A layout) through per-wave LDS slab ----
    #pragma unroll
    for (int nt = 0; nt < 4; ++nt)
      #pragma unroll
      for (int r = 0; r < 8; ++r)
        Ps[wave][r + 8 * half][nt * 16 + l16] = (__bf16)sacc[nt][r];
    // Same-wave LDS ops are in-order; compiler inserts s_wait_dscnt.

    // ---- O += P @ V : K-dim = 64 keys (2 chunks), 8 output d-tiles ----
    #pragma unroll
    for (int c = 0; c < 2; ++c) {
      const __bf16* prow = &Ps[wave][l16][0];
      v16bf pf = cat8(*(const bf16x8*)(prow + c * 32 + half * 8),
                      *(const bf16x8*)(prow + c * 32 + 16 + half * 8));
      #pragma unroll
      for (int dt = 0; dt < 8; ++dt) {
        const __bf16* vrow = &Vt[dt * 16 + l16][c * 32 + 8 * half];
        v16bf vf = cat8(*(const bf16x8*)(vrow), *(const bf16x8*)(vrow + 16));
        oacc[dt] = wmma_bf16(pf, vf, oacc[dt]);
      }
    }
    __syncthreads();   // protect Ks/Vt before next block's staging
  }

  // ---- epilogue: normalize and store f32 [s][HQ][128] ----
  #pragma unroll
  for (int r = 0; r < 8; ++r) {
    float inv = 1.0f / rsum[r];
    int row = q0 + r + 8 * half;
    #pragma unroll
    for (int dt = 0; dt < 8; ++dt) {
      int d = dt * 16 + l16;
      ob[((size_t)row * HQ_N + h) * D_N + d] = oacc[dt][r] * inv;
    }
  }
}

// ---------------------------------------------------------------------------
extern "C" void kernel_launch(void* const* d_in, const int* in_sizes, int n_in,
                              void* d_out, int out_size, void* d_ws, size_t ws_size,
                              hipStream_t stream)
{
  const float* hidden = (const float*)d_in[0];   // [1,2048,4096]
  const float* wq     = (const float*)d_in[1];   // [4096,4096]
  const float* wk     = (const float*)d_in[2];   // [1024,4096]
  const float* wv     = (const float*)d_in[3];   // [1024,4096]
  const float* wo     = (const float*)d_in[4];   // [4096,4096]
  const float* qnw    = (const float*)d_in[5];   // [128]
  const float* knw    = (const float*)d_in[6];   // [128]
  const float* cosb   = (const float*)d_in[7];   // [1,2048,128]
  const float* sinb   = (const float*)d_in[8];   // [1,2048,128]
  const int*   route  = (const int*)d_in[9];     // [1,2048]

  float* qbuf = (float*)d_ws;                                  // 2048*4096
  float* kbuf = qbuf + (size_t)S_LEN * HQ_N * D_N;             // 2048*1024
  float* vbuf = kbuf + (size_t)S_LEN * HKV_N * D_N;            // 2048*1024
  float* abuf = vbuf + (size_t)S_LEN * HKV_N * D_N;            // 2048*4096

  gemm_xwT_bf16wmma<<<dim3((HQ_N * D_N) / 128, S_LEN / 128), 256, 0, stream>>>(
      hidden, wq, qbuf, S_LEN, HQ_N * D_N, HID_N, route);
  gemm_xwT_bf16wmma<<<dim3((HKV_N * D_N) / 128, S_LEN / 128), 256, 0, stream>>>(
      hidden, wk, kbuf, S_LEN, HKV_N * D_N, HID_N, nullptr);
  gemm_xwT_bf16wmma<<<dim3((HKV_N * D_N) / 128, S_LEN / 128), 256, 0, stream>>>(
      hidden, wv, vbuf, S_LEN, HKV_N * D_N, HID_N, nullptr);

  rmsnorm_rope<<<S_LEN * HQ_N, 128, 0, stream>>>(qbuf, qnw, cosb, sinb, HQ_N);
  rmsnorm_rope<<<S_LEN * HKV_N, 128, 0, stream>>>(kbuf, knw, cosb, sinb, HKV_N);

  flash_attn<<<dim3(HQ_N, S_LEN / 64), 128, 0, stream>>>(qbuf, kbuf, vbuf, abuf);

  gemm_xwT_bf16wmma<<<dim3(HID_N / 128, S_LEN / 128), 256, 0, stream>>>(
      abuf, wo, (float*)d_out, S_LEN, HID_N, HID_N, route);
}
